// Random_VQ_18442589569129
// MI455X (gfx1250) — matrix operand order repaired and checked by reading, not compiled, not run on hardware
//
#include <hip/hip_runtime.h>

typedef __attribute__((ext_vector_type(16))) __bf16 v16bf;
typedef __attribute__((ext_vector_type(8)))  float  v8f;

#define DIMX   64
#define CBDIM  32
#define KCB    256
#define HEADS  2
#define MROWS  (8*65536)            /* 524288 rows */
#define ROWS_PER_WG 64
#define NWG    (MROWS/ROWS_PER_WG)  /* 8192 */
#define PSTRIDE 68                  /* padded f32 row stride for P staging */

__device__ __forceinline__ void bsplit(float v, __bf16 &hi, __bf16 &lo) {
  hi = (__bf16)v;
  lo = (__bf16)(v - (float)hi);
}

__global__ __launch_bounds__(128)
void rvq_main(const float* __restrict__ x, const float* __restrict__ rp,
              const float* __restrict__ cb, float* __restrict__ zq,
              int* __restrict__ iout, float* __restrict__ partials)
{
  // Phase-aliased region: W bf16 hi/lo blobs (16 KB) then P staging (17 KB)
  __shared__ __align__(32) unsigned char s_dyn[17408];
  __shared__ __align__(32) __bf16 s_cbhi[16*32*16];   // 16 KB
  __shared__ __align__(32) __bf16 s_cblo[16*32*16];   // 16 KB
  __shared__ float s_cnorm[KCB];
  __shared__ float s_red[128];

  __bf16* s_whi = (__bf16*)s_dyn;
  __bf16* s_wlo = (__bf16*)(s_dyn + 8192);

  const int tid = threadIdx.x;

  // ---- stage rand_projs as pre-swizzled bf16 hi/lo B-operand blobs ----
  // blob (c,t,L): 16 bf16 = B[k=32c+tbl(L>>4,j)][n=16t+(L&15)], B[d][h*32+e]=rp[h][d][e]
  for (int b = tid; b < 256; b += 128) {
    int c = b >> 7;
    int t = (b >> 5) & 3;
    int L = b & 31;
    int n0 = 16*t + (L & 15);
    int hb = L >> 4;
    const float* src = rp + (n0 >> 5)*(DIMX*CBDIM) + (n0 & 31);
    #pragma unroll
    for (int j = 0; j < 16; ++j) {
      int d = 32*c + ((j < 8) ? 8*hb + j : 16 + 8*hb + (j - 8));
      __bf16 h, l; bsplit(src[d*CBDIM], h, l);
      s_whi[b*16 + j] = h; s_wlo[b*16 + j] = l;
    }
  }
  // ---- stage codebook^T as bf16 hi/lo B-operand blobs ----
  // blob (kt,L): 16 bf16 = codebook[16kt+(L&15)][e=tbl(L>>4,j)], j=0..15
  for (int b = tid; b < 512; b += 128) {
    int kt = b >> 5, L = b & 31;
    #pragma unroll
    for (int j = 0; j < 16; ++j) {
      __bf16 h, l; bsplit(cb[(16*kt + (L & 15))*CBDIM + ((j < 8) ? 8*(L >> 4) + j : 16 + 8*(L >> 4) + (j - 8))], h, l);
      s_cbhi[b*16 + j] = h; s_cblo[b*16 + j] = l;
    }
  }
  // ---- codeword squared norms ----
  for (int k = tid; k < KCB; k += 128) {
    const float* cr = cb + k*CBDIM;
    float s = 0.f;
    #pragma unroll
    for (int e = 0; e < CBDIM; ++e) s += cr[e]*cr[e];
    s_cnorm[k] = s;
  }
  __syncthreads();

  const int wave = tid >> 5, lane = tid & 31;
  const int m  = lane & 15;       // row within slab (A-layout)
  const int hh = lane >> 4;       // K-half selector
  const int off = 8*hh;
  const int slab = blockIdx.x*ROWS_PER_WG + wave*16;

  // ---- GEMM1: P = x @ W, split-bf16 (hi*hi + hi*lo + lo*hi), f32 accum ----
  const float* xr = x + (size_t)(slab + m)*DIMX;
  v16bf Ahi[2], Alo[2];
  #pragma unroll
  for (int c = 0; c < 2; ++c) {
    #pragma unroll
    for (int part = 0; part < 2; ++part) {   // K = 32c + 16*part + off + (0..7)
      const float4* p4 = (const float4*)(xr + 32*c + 16*part + off);
      float4 a = p4[0], b2 = p4[1];
      const float v[8] = {a.x,a.y,a.z,a.w,b2.x,b2.y,b2.z,b2.w};
      #pragma unroll
      for (int i = 0; i < 8; ++i) {
        __bf16 h, l; bsplit(v[i], h, l);
        Ahi[c][8*part + i] = h; Alo[c][8*part + i] = l;
      }
    }
  }
  const v8f zero8 = {0.f,0.f,0.f,0.f,0.f,0.f,0.f,0.f};
  v8f pacc[4];
  #pragma unroll
  for (int t = 0; t < 4; ++t) {
    v8f acc = zero8;
    #pragma unroll
    for (int c = 0; c < 2; ++c) {
      const v16bf Bh = *(const v16bf*)(s_whi + ((c*4 + t)*32 + lane)*16);
      const v16bf Bl = *(const v16bf*)(s_wlo + ((c*4 + t)*32 + lane)*16);
      acc = __builtin_amdgcn_wmma_f32_16x16x32_bf16(false, Ahi[c], false, Bh, (short)0, acc, false, false);
      acc = __builtin_amdgcn_wmma_f32_16x16x32_bf16(false, Ahi[c], false, Bl, (short)0, acc, false, false);
      acc = __builtin_amdgcn_wmma_f32_16x16x32_bf16(false, Alo[c], false, Bh, (short)0, acc, false, false);
    }
    pacc[t] = acc;
  }
  __syncthreads();   // all waves done with W region -> reuse as P staging

  float* ps = (float*)s_dyn + wave*16*PSTRIDE;   // wave-private 16x68 f32
  #pragma unroll
  for (int t = 0; t < 4; ++t)
    #pragma unroll
    for (int r = 0; r < 8; ++r)
      ps[(r + 8*hh)*PSTRIDE + 16*t + m] = pacc[t][r];

  // ---- GEMM2 + argmin per head ----
  float lossAcc = 0.f;
  #pragma unroll
  for (int h = 0; h < HEADS; ++h) {
    const float* prow = ps + m*PSTRIDE + h*CBDIM;
    v16bf Phi, Plo;
    #pragma unroll
    for (int part = 0; part < 2; ++part) {
      const float4* p4 = (const float4*)(prow + 16*part + off);
      float4 a = p4[0], b2 = p4[1];
      const float v[8] = {a.x,a.y,a.z,a.w,b2.x,b2.y,b2.z,b2.w};
      #pragma unroll
      for (int i = 0; i < 8; ++i) {
        lossAcc += v[i]*v[i];                  // Sum ||p||^2 (loss term A)
        __bf16 hx, lx; bsplit(v[i], hx, lx);
        Phi[8*part + i] = hx; Plo[8*part + i] = lx;
      }
    }
    float best[8]; int bidx[8];
    #pragma unroll
    for (int r = 0; r < 8; ++r) { best[r] = 3.402823e38f; bidx[r] = 0; }

    for (int kt = 0; kt < 16; ++kt) {
      const int kk = 16*kt + m;
      const float cn = s_cnorm[kk];
      const v16bf Bh = *(const v16bf*)(s_cbhi + (kt*32 + lane)*16);
      const v16bf Bl = *(const v16bf*)(s_cblo + (kt*32 + lane)*16);
      v8f acc = zero8;
      acc = __builtin_amdgcn_wmma_f32_16x16x32_bf16(false, Phi, false, Bh, (short)0, acc, false, false);
      acc = __builtin_amdgcn_wmma_f32_16x16x32_bf16(false, Phi, false, Bl, (short)0, acc, false, false);
      acc = __builtin_amdgcn_wmma_f32_16x16x32_bf16(false, Plo, false, Bh, (short)0, acc, false, false);
      #pragma unroll
      for (int r = 0; r < 8; ++r) {
        float d2 = cn - 2.f*acc[r];            // + ||p||^2 is row-constant
        if (d2 < best[r]) { best[r] = d2; bidx[r] = kk; }
      }
    }
    // reduce (val,idx) across the 16 lanes holding the same rows
    #pragma unroll
    for (int r = 0; r < 8; ++r) {
      #pragma unroll
      for (int sh = 8; sh >= 1; sh >>= 1) {
        float ov = __shfl_xor(best[r], sh, 32);
        int   oi = __shfl_xor(bidx[r], sh, 32);
        if (ov < best[r] || (ov == best[r] && oi < bidx[r])) { best[r] = ov; bidx[r] = oi; }
      }
    }
    float bs = 0.f;
    #pragma unroll
    for (int r = 0; r < 8; ++r) bs += best[r];
    lossAcc += bs * 0.0625f;                   // group of 16 lanes adds same sum

    // z_q = codebook[idx]  (exact f32 gather, coalesced float2 per lane)
    #pragma unroll
    for (int r = 0; r < 8; ++r) {
      const int row = slab + r + 8*hh;
      const float2 q2 = *(const float2*)(cb + bidx[r]*CBDIM + 2*m);
      *(float2*)(zq + (size_t)row*(HEADS*CBDIM) + h*CBDIM + 2*m) = q2;
    }
    if (lane == 0) {
      #pragma unroll
      for (int r = 0; r < 8; ++r) iout[(slab + r)*HEADS + h] = bidx[r];
    } else if (lane == 16) {
      #pragma unroll
      for (int r = 0; r < 8; ++r) iout[(slab + 8 + r)*HEADS + h] = bidx[r];
    }
  }

  // ---- deterministic per-WG loss reduction ----
  s_red[tid] = lossAcc;
  __syncthreads();
  #pragma unroll
  for (int st = 64; st >= 1; st >>= 1) {
    if (tid < st) s_red[tid] += s_red[tid + st];
    __syncthreads();
  }
  if (tid == 0) partials[blockIdx.x] = s_red[0];
}

__global__ __launch_bounds__(256)
void rvq_loss_reduce(const float* __restrict__ partials, float* __restrict__ lout)
{
  __shared__ float sr[256];
  const int t = threadIdx.x;
  float s = 0.f;
  for (int i = t; i < NWG; i += 256) s += partials[i];   // fixed order
  sr[t] = s;
  __syncthreads();
  #pragma unroll
  for (int st = 128; st >= 1; st >>= 1) {
    if (t < st) sr[t] += sr[t + st];
    __syncthreads();
  }
  if (t == 0) lout[0] = sr[0] * (1.0f / (float)((size_t)MROWS*HEADS*CBDIM));
}

extern "C" void kernel_launch(void* const* d_in, const int* in_sizes, int n_in,
                              void* d_out, int out_size, void* d_ws, size_t ws_size,
                              hipStream_t stream) {
  const float* x  = (const float*)d_in[0];
  const float* rp = (const float*)d_in[1];
  const float* cb = (const float*)d_in[2];
  float* out  = (float*)d_out;
  float* zq   = out;                                        // 33,554,432 f32
  int*   iout = (int*)(out + (size_t)MROWS*HEADS*CBDIM);    // 1,048,576 i32
  float* lout = out + (size_t)MROWS*HEADS*CBDIM + (size_t)MROWS*HEADS;
  float* partials = (float*)d_ws;                           // 8192 f32

  rvq_main<<<NWG, 128, 0, stream>>>(x, rp, cb, zq, iout, partials);
  rvq_loss_reduce<<<1, 256, 0, stream>>>(partials, lout);
}
